// EagleSparseMoeBlock_420906795809
// MI455X (gfx1250) — matrix operand order repaired and checked by reading, not compiled, not run on hardware
//
#include <hip/hip_runtime.h>
#include <hip/hip_bf16.h>
#include <math.h>

// MI455X / gfx1250, wave32. bf16 WMMA 16x16x32 (f32 accum) + TDM async weight staging.

typedef __bf16 bf16_t;
typedef __attribute__((ext_vector_type(16))) __bf16 v16bf;
typedef __attribute__((ext_vector_type(8)))  float  v8f;
typedef __attribute__((ext_vector_type(4)))  unsigned int ui32x4;
typedef __attribute__((ext_vector_type(8)))  int i32x8;
typedef __attribute__((ext_vector_type(4)))  int i32x4;

#define T_TOK   8192
#define HDIM    1024
#define FDIM    4096
#define NEXP    8
#define TM      64     // tokens per block
#define HC      256    // output columns per block
#define FC      64     // F chunk
#define KC      128    // K chunk for GEMM1
#define XPAD    8
#define WPAD    8

// A-matrix fragment (16x32 bf16) per ISA layout:
//   lane: m = lane&15, half = lane>>4 ; element j -> k = (j>>3)*16 + half*8 + (j&7)
__device__ __forceinline__ v16bf load_afrag(const bf16_t* __restrict__ row, int kbase, int hl) {
  v16bf a;
  const bf16_t* p0 = row + kbase + hl * 8;
  const bf16_t* p1 = row + kbase + 16 + hl * 8;
#pragma unroll
  for (int j = 0; j < 8; ++j) { a[j] = p0[j]; a[j + 8] = p1[j]; }
  return a;
}

// B-matrix fragment (32x16 bf16): lane: n = lane&15, half = lane>>4;
//   element j -> k = (j>>3)*16 + half*8 + (j&7).  W stored transposed in LDS as [n][k].
__device__ __forceinline__ v16bf load_bfrag(const bf16_t* __restrict__ row, int kbase, int hl) {
  v16bf b;
  const bf16_t* p0 = row + kbase + hl * 8;
  const bf16_t* p1 = row + kbase + 16 + hl * 8;
#pragma unroll
  for (int j = 0; j < 8; ++j) { b[j] = p0[j]; b[j + 8] = p1[j]; }
  return b;
}

// Tensor Data Mover: async DMA of a [KC rows x FC cols] f32 tile (row stride FDIM)
// from global memory into LDS at lds_off.  2-D tensor -> descriptor groups 2/3 zero.
__device__ __forceinline__ void tdm_load_w_chunk(const float* gptr, unsigned int lds_off) {
  unsigned long long ga = (unsigned long long)(uintptr_t)gptr;
  ui32x4 g0;
  g0[0] = 1u;                                                // count=1, user descriptor
  g0[1] = lds_off;                                           // lds_addr (bytes)
  g0[2] = (unsigned int)(ga & 0xFFFFFFFFull);                // global_addr[31:0]
  g0[3] = (unsigned int)((ga >> 32) & 0x1FFFFFFull)          // global_addr[56:32]
          | (0x2u << 30);                                    // type = 2 ("image")
  i32x8 g1;
  g1[0] = (int)(2u << 16);          // workgroup_mask=0, data_size=2 (4 bytes)
  g1[1] = (int)((unsigned)FC << 16);   // tensor_dim0[15:0] = 64  (bits 63:48)
  g1[2] = (int)((unsigned)KC << 16);   // tensor_dim1[15:0] = 128 (bits 95:80)
  g1[3] = (int)((unsigned)FC << 16);   // tile_dim0 = 64          (bits 127:112)
  g1[4] = (int)KC;                     // tile_dim1 = 128, tile_dim2 = 0
  g1[5] = (int)FDIM;                   // tensor_dim0_stride[31:0] = 4096
  g1[6] = 0;
  g1[7] = 0;
  i32x4 z4 = {0, 0, 0, 0};
  i32x8 z8 = {0, 0, 0, 0, 0, 0, 0, 0};
  // 6-arg form (clang-23 / therock-10.0 headers)
  __builtin_amdgcn_tensor_load_to_lds(g0, g1, z4, z4, z8, 0);
}

__global__ __launch_bounds__(256, 1)
void moe_fused_kernel(const float* __restrict__ X,
                      const float* __restrict__ Wg,
                      const float* __restrict__ W1,
                      const float* __restrict__ W3,
                      const float* __restrict__ W2,
                      float* __restrict__ Out,
                      float* __restrict__ Logits)
{
  __shared__ __align__(16) bf16_t sX[TM][HDIM + XPAD];     // 132,096 B
  __shared__ __align__(16) bf16_t sW1[FC][KC + WPAD];      //  17,408 B
  __shared__ __align__(16) bf16_t sW3[FC][KC + WPAD];      //  17,408 B
  __shared__ __align__(16) bf16_t sW2[HC][FC + WPAD];      //  36,864 B
  __shared__ __align__(16) bf16_t sAct[TM][FC + WPAD];     //   9,216 B
  __shared__ float sCW[TM][NEXP];                          //   2,048 B
  __shared__ __align__(16) float sW1f[KC * FC];            //  32,768 B (TDM staging)
  __shared__ __align__(16) float sW3f[KC * FC];            //  32,768 B (TDM staging)
  // total ~280.6 KB of the 320 KB WGP LDS

  const int tid  = threadIdx.x;
  const int lane = tid & 31;
  const int wv   = tid >> 5;
  const int hl   = lane >> 4;   // half-wave select
  const int ln16 = lane & 15;
  const int tm0  = blockIdx.x * TM;
  const int hc0  = blockIdx.y * HC;

  const unsigned int off_w1f = (unsigned int)(uintptr_t)(void*)sW1f;
  const unsigned int off_w3f = (unsigned int)(uintptr_t)(void*)sW3f;

  // ---- Phase 0: X tile [64 x 1024] f32 -> bf16 LDS ----
  for (int i = tid; i < TM * (HDIM / 4); i += 256) {
    int r  = i >> 8;            // HDIM/4 == 256
    int c4 = i & 255;
    const float4 v = reinterpret_cast<const float4*>(X)[(size_t)(tm0 + r) * (HDIM / 4) + c4];
    sX[r][c4 * 4 + 0] = (bf16_t)v.x;
    sX[r][c4 * 4 + 1] = (bf16_t)v.y;
    sX[r][c4 * 4 + 2] = (bf16_t)v.z;
    sX[r][c4 * 4 + 3] = (bf16_t)v.w;
  }
  __syncthreads();

  // ---- Phase 1: router logits + softmax + top-2 combine weights ----
  {
    const int tok  = tid >> 2;  // 4 threads per token
    const int part = tid & 3;
    float acc[NEXP];
#pragma unroll
    for (int e = 0; e < NEXP; ++e) acc[e] = 0.f;
    for (int h = part * 256; h < part * 256 + 256; ++h) {
      float xv = (float)sX[tok][h];
      const float* wg = Wg + (size_t)h * NEXP;
#pragma unroll
      for (int e = 0; e < NEXP; ++e) acc[e] = fmaf(xv, wg[e], acc[e]);
    }
#pragma unroll
    for (int e = 0; e < NEXP; ++e) {
      acc[e] += __shfl_xor(acc[e], 1, 32);
      acc[e] += __shfl_xor(acc[e], 2, 32);
    }
    if (part == 0) {
      float mx = acc[0];
#pragma unroll
      for (int e = 1; e < NEXP; ++e) mx = fmaxf(mx, acc[e]);
      float p[NEXP];
#pragma unroll
      for (int e = 0; e < NEXP; ++e) p[e] = __expf(acc[e] - mx);
      int i1 = 0;
#pragma unroll
      for (int e = 1; e < NEXP; ++e) if (p[e] > p[i1]) i1 = e;
      int i2 = (i1 == 0) ? 1 : 0;
#pragma unroll
      for (int e = 0; e < NEXP; ++e) if (e != i1 && p[e] > p[i2]) i2 = e;
      float denom = p[i1] + p[i2];
#pragma unroll
      for (int e = 0; e < NEXP; ++e)
        sCW[tok][e] = (e == i1) ? p[i1] / denom : ((e == i2) ? p[i2] / denom : 0.f);
      if (blockIdx.y == 0) {
        float* lg = Logits + (size_t)(tm0 + tok) * NEXP;
#pragma unroll
        for (int e = 0; e < NEXP; ++e) lg[e] = acc[e];
      }
    }
  }
  __syncthreads();

  // ---- Phase 2: fused expert loop ----
  const int mt    = wv & 3;           // M tile (16 rows) for this wave
  const int g1nt0 = (wv >> 2) * 2;    // GEMM1: 2 N-tiles of the 64-col F chunk
  const int g2nt0 = (wv >> 2) * 8;    // GEMM2: 8 N-tiles of the 256-col H chunk
  const bf16_t* aRowX   = &sX[mt * 16 + ln16][0];
  const bf16_t* aRowAct = &sAct[mt * 16 + ln16][0];

  const v8f vzero = {0.f, 0.f, 0.f, 0.f, 0.f, 0.f, 0.f, 0.f};
  v8f outAcc[8];
#pragma unroll
  for (int t = 0; t < 8; ++t) outAcc[t] = vzero;

  for (int e = 0; e < NEXP; ++e) {
    const float* w1e = W1 + (size_t)e * HDIM * FDIM;
    const float* w3e = W3 + (size_t)e * HDIM * FDIM;
    const float* w2e = W2 + (size_t)e * FDIM * HDIM;

    for (int fc = 0; fc < FDIM; fc += FC) {
      v8f p1[2] = {vzero, vzero};
      v8f p3[2] = {vzero, vzero};

      // Kick off async TDM DMA of the first K-chunk of W1/W3 (wave 0 owns TENSORcnt)
      __syncthreads();
      if (wv == 0) {
        tdm_load_w_chunk(w1e + (size_t)0 * FDIM + fc, off_w1f);
        tdm_load_w_chunk(w3e + (size_t)0 * FDIM + fc, off_w3f);
      }

      // GEMM1 / GEMM3: P1 = X @ W1[:, fc:fc+64], P3 = X @ W3[:, fc:fc+64]
      for (int kb = 0; kb < HDIM; kb += KC) {
        if (wv == 0) __builtin_amdgcn_s_wait_tensorcnt(0);  // DMA of this chunk done
        __syncthreads();   // staging visible to all; previous wmma done with sW1/sW3

        // convert f32 staging (k-major [k][n]) -> bf16 transposed [n][k]
        for (int i = tid; i < (KC * FC) / 4; i += 256) {
          int k  = i >> 4;
          int n4 = (i & 15) * 4;
          const float4 a1 = *reinterpret_cast<const float4*>(sW1f + k * FC + n4);
          const float4 a3 = *reinterpret_cast<const float4*>(sW3f + k * FC + n4);
          sW1[n4 + 0][k] = (bf16_t)a1.x; sW1[n4 + 1][k] = (bf16_t)a1.y;
          sW1[n4 + 2][k] = (bf16_t)a1.z; sW1[n4 + 3][k] = (bf16_t)a1.w;
          sW3[n4 + 0][k] = (bf16_t)a3.x; sW3[n4 + 1][k] = (bf16_t)a3.y;
          sW3[n4 + 2][k] = (bf16_t)a3.z; sW3[n4 + 3][k] = (bf16_t)a3.w;
        }
        __syncthreads();   // bf16 tiles ready; staging consumed

        // prefetch next K-chunk via TDM, overlapping with the WMMA work below
        if (wv == 0 && kb + KC < HDIM) {
          tdm_load_w_chunk(w1e + (size_t)(kb + KC) * FDIM + fc, off_w1f);
          tdm_load_w_chunk(w3e + (size_t)(kb + KC) * FDIM + fc, off_w3f);
        }

#pragma unroll
        for (int kk = 0; kk < 4; ++kk) {
          v16bf a = load_afrag(aRowX, kb + kk * 32, hl);
#pragma unroll
          for (int t = 0; t < 2; ++t) {
            v16bf b1 = load_bfrag(&sW1[(g1nt0 + t) * 16 + ln16][0], kk * 32, hl);
            p1[t] = __builtin_amdgcn_wmma_f32_16x16x32_bf16(false, a, false, b1,
                                                            (short)0, p1[t], false, false);
            v16bf b3 = load_bfrag(&sW3[(g1nt0 + t) * 16 + ln16][0], kk * 32, hl);
            p3[t] = __builtin_amdgcn_wmma_f32_16x16x32_bf16(false, a, false, b3,
                                                            (short)0, p3[t], false, false);
          }
        }
      }

      // act = silu(P1) * P3, scaled by this expert's routing weight (folds combine
      // into the second GEMM so outAcc can sum across experts with no atomics)
      __syncthreads();
#pragma unroll
      for (int t = 0; t < 2; ++t) {
        int n = (g1nt0 + t) * 16 + ln16;
#pragma unroll
        for (int v = 0; v < 8; ++v) {
          int m = mt * 16 + hl * 8 + v;   // C/D layout: lanes 0-15 -> M 0-7, 16-31 -> M 8-15
          float g = p1[t][v];
          float u = p3[t][v];
          float act = (g / (1.f + __expf(-g))) * u * sCW[m][e];
          sAct[m][n] = (bf16_t)act;
        }
      }
      // stage W2 chunk [f=fc..fc+63][hc0..hc0+255] transposed into sW2[n][f] (b128 loads)
      for (int i = tid; i < (HC * FC) / 4; i += 256) {
        int f  = i >> 6;          // 0..63
        int n4 = (i & 63) * 4;    // 0..252, coalesced along H
        const float4 v = *reinterpret_cast<const float4*>(
            w2e + (size_t)(fc + f) * HDIM + hc0 + n4);
        sW2[n4 + 0][f] = (bf16_t)v.x; sW2[n4 + 1][f] = (bf16_t)v.y;
        sW2[n4 + 2][f] = (bf16_t)v.z; sW2[n4 + 3][f] = (bf16_t)v.w;
      }
      __syncthreads();

      // GEMM2: outAcc += act @ W2[fc:fc+64, hc0:hc0+256]
#pragma unroll
      for (int kk = 0; kk < 2; ++kk) {
        v16bf a = load_afrag(aRowAct, kk * 32, hl);
#pragma unroll
        for (int t = 0; t < 8; ++t) {
          v16bf b = load_bfrag(&sW2[(g2nt0 + t) * 16 + ln16][0], kk * 32, hl);
          outAcc[t] = __builtin_amdgcn_wmma_f32_16x16x32_bf16(false, a, false, b,
                                                              (short)0, outAcc[t], false, false);
        }
      }
    }
  }

  // ---- Phase 3: store the 64 x 256 output tile ----
#pragma unroll
  for (int t = 0; t < 8; ++t) {
#pragma unroll
    for (int v = 0; v < 8; ++v) {
      int m = mt * 16 + hl * 8 + v;
      int n = (g2nt0 + t) * 16 + ln16;
      Out[(size_t)(tm0 + m) * HDIM + hc0 + n] = outAcc[t][v];
    }
  }
}

extern "C" void kernel_launch(void* const* d_in, const int* in_sizes, int n_in,
                              void* d_out, int out_size, void* d_ws, size_t ws_size,
                              hipStream_t stream) {
  (void)in_sizes; (void)n_in; (void)out_size; (void)d_ws; (void)ws_size;
  const float* X  = (const float*)d_in[0];
  const float* Wg = (const float*)d_in[1];
  const float* W1 = (const float*)d_in[2];
  const float* W3 = (const float*)d_in[3];
  const float* W2 = (const float*)d_in[4];
  float* Out    = (float*)d_out;
  float* Logits = Out + (size_t)T_TOK * HDIM;   // tuple output: [out | router_logits]
  dim3 grid(T_TOK / TM, HDIM / HC);             // 128 x 4 blocks
  moe_fused_kernel<<<grid, 256, 0, stream>>>(X, Wg, W1, W3, W2, Out, Logits);
}
